// DSVDD_45973329936668
// MI455X (gfx1250) — compile-verified, matrix operand order checked by reading, not run on hardware
//
#include <hip/hip_runtime.h>
#include <hip/hip_bf16.h>
#include <math.h>

typedef __attribute__((ext_vector_type(16))) __bf16 v16bf;
typedef __attribute__((ext_vector_type(8)))  float  v8f;
typedef __attribute__((ext_vector_type(4)))  unsigned int uint32x4;
typedef __attribute__((ext_vector_type(8)))  int  int32x8;
typedef __attribute__((ext_vector_type(4)))  int  int32x4;

// ---------------- problem constants ----------------
#define BATCH   8
#define HW      3136            // 56*56
#define MROWS   (BATCH*HW)      // 25088
#define K1PAD   1824            // 1794 padded to 57*32
#define K1REAL  1794
#define NPHI    1792
#define NCENT   3136
#define K2      1792            // 56*32

// ---------------- workspace layout (bytes) ----------------
#define OFF_WB   ((size_t)0)                       // bf16 1792*1824
#define OFF_CT   ((size_t)6537216)                 // bf16 3136*1792
#define OFF_CN   ((size_t)17776640)                // f32  3136
#define OFF_AP1  ((size_t)17789184)                // f32  8*256*56*56
#define OFF_AP2  ((size_t)43479296)                // f32  8*512*28*28
#define OFF_AP3  ((size_t)56324352)                // f32  8*1024*14*14
#define OFF_X    ((size_t)62746880)                // bf16 25088*1824
#define OFF_PHI  ((size_t)154267904)               // bf16 25088*1792
#define OFF_RN   ((size_t)244183296)               // f32  25088

// ---------------- Tensor Data Mover: 2-D bf16 tile -> LDS ----------------
// D# per CDNA5 ISA ch.8: group0 = {flags, lds_addr, global_addr lo, global_addr hi | type=2}
// group1 = {mask/data_size, tensor_dim0/1, tile_dim0/1/2, tensor_dim0_stride, tensor_dim1_stride}
// data_size=1 (2-byte elements). Issued by one wave; EXEC is ignored by TDM.
__device__ __forceinline__ void tdm_load_tile_bf16(unsigned lds_off, const void* gptr,
                                                   unsigned dim0, unsigned rows,
                                                   unsigned stride_elems) {
    unsigned long long ga = (unsigned long long)gptr;
    uint32x4 g0;
    g0[0] = 1u;                                             // count=1, user descriptor
    g0[1] = lds_off;                                        // LDS byte address
    g0[2] = (unsigned)(ga & 0xffffffffu);                   // global_addr[31:0]
    g0[3] = (unsigned)((ga >> 32) & 0x01ffffffu) | 0x80000000u;  // [56:32] | type=2
    int32x8 g1;
    g1[0] = 0x00010000;                                     // data_size=1 (2B), wg_mask=0
    g1[1] = (int)((dim0 & 0xffffu) << 16);                  // tensor_dim0[15:0]
    g1[2] = (int)(((dim0 >> 16) & 0xffffu) | ((rows & 0xffffu) << 16)); // dim0 hi | tensor_dim1 lo
    g1[3] = (int)((dim0 & 0xffffu) << 16);                  // tensor_dim1 hi=0 | tile_dim0
    g1[4] = (int)(rows & 0xffffu);                          // tile_dim1=rows, tile_dim2=0
    g1[5] = (int)stride_elems;                              // tensor_dim0_stride lo32
    g1[6] = 0;                                              // stride hi | tensor_dim1_stride lo
    g1[7] = 0;
    int32x4 z4 = {};
#if defined(__clang_major__) && __clang_major__ >= 23
    int32x8 z8 = {};
    __builtin_amdgcn_tensor_load_to_lds(g0, g1, z4, z4, z8, 0);
#else
    __builtin_amdgcn_tensor_load_to_lds(g0, g1, z4, z4, 0);
#endif
}

// ---------------- small helpers ----------------
__device__ __forceinline__ void ins3(float d, float& v0, float& v1, float& v2) {
    float M0 = fmaxf(d, v0);
    v0 = fminf(d, v0);
    float M1 = fmaxf(M0, v1);
    v1 = fminf(M0, v1);
    v2 = fminf(M1, v2);
}

__device__ __forceinline__ void merge3(float& a0, float& a1, float& a2,
                                       float b0, float b1, float b2) {
    float c0 = fminf(a0, b0);
    float x  = fmaxf(a0, b0);
    float y  = fminf(a1, b1);
    float z  = fmaxf(a1, b1);
    float w  = fminf(a2, b2);
    float c1 = fminf(x, y);
    float c2 = fminf(fmaxf(x, y), fminf(z, w));
    a0 = c0; a1 = c1; a2 = c2;
}

// ============== prep: W -> bf16 padded (B-fragment layout: row=o, K contiguous) ==============
__global__ void __launch_bounds__(256) k_prep_W(const float* __restrict__ W,
                                                __bf16* __restrict__ Wb) {
    int idx = blockIdx.x * 256 + threadIdx.x;
    if (idx >= NPHI * K1PAD) return;
    int o = idx / K1PAD, k = idx % K1PAD;
    float v = (k < K1REAL) ? W[(size_t)o * K1REAL + k] : 0.0f;
    Wb[idx] = (__bf16)v;
}

// ============== prep: C (1792x3136) -> Ct bf16 (3136x1792), tiled LDS transpose ==============
__global__ void __launch_bounds__(256) k_transpose_C(const float* __restrict__ C,
                                                     __bf16* __restrict__ Ct) {
    __shared__ float tile[32][33];
    int tx = threadIdx.x & 31, ty = threadIdx.x >> 5;     // 32 x 8
    int nB = blockIdx.x * 32, kB = blockIdx.y * 32;
#pragma unroll
    for (int j = 0; j < 4; ++j)
        tile[ty + j * 8][tx] = C[(size_t)(kB + ty + j * 8) * NCENT + nB + tx];
    __syncthreads();
#pragma unroll
    for (int j = 0; j < 4; ++j)
        Ct[(size_t)(nB + ty + j * 8) * K2 + kB + tx] = (__bf16)tile[tx][ty + j * 8];
}

// ============== prep: colnorm[n] = sum_k bf16(C[k][n])^2 ==============
__global__ void __launch_bounds__(256) k_colnorm(const float* __restrict__ C,
                                                 float* __restrict__ cn) {
    int n = blockIdx.x * 256 + threadIdx.x;
    if (n >= NCENT) return;
    float s = 0.0f;
    for (int k = 0; k < K2; ++k) {
        float v = (float)(__bf16)C[(size_t)k * NCENT + n];
        s += v * v;
    }
    cn[n] = s;
}

// ============== 3x3 avg pool (stride1, zero pad, /9) ==============
__global__ void __launch_bounds__(256) k_avgpool(const float* __restrict__ in,
                                                 float* __restrict__ out,
                                                 int C, int H) {
    int total = BATCH * C * H * H;
    int idx = blockIdx.x * 256 + threadIdx.x;
    if (idx >= total) return;
    int x = idx % H, y = (idx / H) % H, bc = idx / (H * H);
    const float* base = in + (size_t)bc * H * H;
    float s = 0.0f;
#pragma unroll
    for (int dy = -1; dy <= 1; ++dy) {
        int yy = y + dy;
        if (yy < 0 || yy >= H) continue;
#pragma unroll
        for (int dx = -1; dx <= 1; ++dx) {
            int xx = x + dx;
            if (xx < 0 || xx >= H) continue;
            s += base[yy * H + xx];
        }
    }
    out[idx] = s * (1.0f / 9.0f);
}

// ============== build descriptor X bf16 (25088 x 1824) ==============
__device__ __forceinline__ float bilin(const float* __restrict__ plane, int S,
                                       float sy, float sx) {
    int iy0 = (int)floorf(sy);
    int ix0 = (int)floorf(sx);
    float fy = sy - (float)iy0;
    float fx = sx - (float)ix0;
    int ya = min(max(iy0, 0), S - 1), yb = min(max(iy0 + 1, 0), S - 1);
    int xa = min(max(ix0, 0), S - 1), xb = min(max(ix0 + 1, 0), S - 1);
    float v00 = plane[ya * S + xa], v01 = plane[ya * S + xb];
    float v10 = plane[yb * S + xa], v11 = plane[yb * S + xb];
    float t0 = v00 + fx * (v01 - v00);
    float t1 = v10 + fx * (v11 - v10);
    return t0 + fy * (t1 - t0);
}

__global__ void __launch_bounds__(256) k_build_X(const float* __restrict__ ap1,
                                                 const float* __restrict__ ap2,
                                                 const float* __restrict__ ap3,
                                                 __bf16* __restrict__ Xb) {
    int idx = blockIdx.x * 256 + threadIdx.x;
    if (idx >= MROWS * K1PAD) return;
    int c = idx % K1PAD;
    int m = idx / K1PAD;
    int b = m / HW, p = m % HW;
    int y = p / 56, x = p % 56;
    float val;
    if (c < 256) {
        val = ap1[((size_t)(b * 256 + c) * HW) + p];
    } else if (c < 768) {
        const float* plane = ap2 + (size_t)(b * 512 + (c - 256)) * 28 * 28;
        val = bilin(plane, 28, (float)y * 0.5f - 0.25f, (float)x * 0.5f - 0.25f);
    } else if (c < 1792) {
        const float* plane = ap3 + (size_t)(b * 1024 + (c - 768)) * 14 * 14;
        val = bilin(plane, 14, (float)y * 0.25f - 0.375f, (float)x * 0.25f - 0.375f);
    } else if (c == 1792) {
        val = (float)x * (2.0f / 55.0f) - 1.0f;
    } else if (c == 1793) {
        val = (float)y * (2.0f / 55.0f) - 1.0f;
    } else {
        val = 0.0f;       // K padding
    }
    Xb[idx] = (__bf16)val;
}

// ============== GEMM1: phi = X * Wb^T + b ==============
// One block per 16-row M-tile; X rows TDM-staged once into LDS; 4 waves sweep 112 N-tiles.
__global__ void __launch_bounds__(128) k_gemm1(const __bf16* __restrict__ Xb,
                                               const __bf16* __restrict__ Wb,
                                               const float*  __restrict__ bias,
                                               __bf16* __restrict__ phi) {
    __shared__ __align__(16) __bf16 sX[16 * K1PAD];     // 58368 B, LDS offset 0
    int tid = threadIdx.x, wave = tid >> 5, lane = tid & 31;
    int mbase = blockIdx.x * 16;

    if (wave == 0) {
        tdm_load_tile_bf16(0u, Xb + (size_t)mbase * K1PAD, K1PAD, 16, K1PAD);
        __builtin_amdgcn_s_wait_tensorcnt(0);
    }
    // The TDM writes sX behind the compiler's back; without a visible writer the
    // A-fragment ds_loads get folded as reads of never-stored memory. Provide a
    // never-taken store and an opaque clobber of the LDS tile.
    if (Xb == nullptr) sX[tid] = (__bf16)0.0f;
    asm volatile("" : : "v"(sX) : "memory");
    __syncthreads();

    int lrow = lane & 15;
    int hi = lane >> 4;
    int ksel = hi << 4;                                 // 0 or 16
    const __bf16* arow = sX + lrow * K1PAD + ksel;

    for (int nt = wave; nt < 112; nt += 4) {
        const __bf16* brow = Wb + (size_t)(nt * 16 + lrow) * K1PAD + ksel;
        v8f acc = {};
#pragma unroll 4
        for (int kc = 0; kc < 57; ++kc) {
            v16bf a = *(const v16bf*)(arow + kc * 32);
            v16bf bm = *(const v16bf*)(brow + kc * 32);
            acc = __builtin_amdgcn_wmma_f32_16x16x32_bf16(false, a, false, bm,
                                                          (short)0, acc, false, false);
        }
        float bv = bias[nt * 16 + lrow];
#pragma unroll
        for (int i = 0; i < 8; ++i) {
            int r = mbase + hi * 8 + i;
            phi[(size_t)r * NPHI + nt * 16 + lrow] = (__bf16)(acc[i] + bv);
        }
    }
}

// ============== rownorm[m] = sum_o phi[m][o]^2  (one wave per row) ==============
__global__ void __launch_bounds__(128) k_rownorm(const __bf16* __restrict__ phi,
                                                 float* __restrict__ rn) {
    int wave = threadIdx.x >> 5, lane = threadIdx.x & 31;
    int row = blockIdx.x * 4 + wave;
    const unsigned* p = (const unsigned*)(phi + (size_t)row * NPHI);   // 896 dwords
    float s = 0.0f;
    for (int i = lane; i < 896; i += 32) {
        unsigned u = p[i];
        float a = __uint_as_float(u << 16);
        float b = __uint_as_float(u & 0xffff0000u);
        s += a * a + b * b;
    }
#pragma unroll
    for (int m = 1; m < 32; m <<= 1) s += __shfl_xor(s, m, 32);
    if (lane == 0) rn[row] = s;
}

// ============== GEMM2 fused: dist + sqrt + top-3 + softmin score ==============
__global__ void __launch_bounds__(128) k_gemm2_topk(const __bf16* __restrict__ phi,
                                                    const __bf16* __restrict__ Ct,
                                                    const float*  __restrict__ rownorm,
                                                    const float*  __restrict__ colnorm,
                                                    float* __restrict__ out) {
    __shared__ __align__(16) __bf16 sPhi[16 * K2];      // 57344 B, LDS offset 0
    __shared__ float sRed[4][16][4];
    int tid = threadIdx.x, wave = tid >> 5, lane = tid & 31;
    int mbase = blockIdx.x * 16;

    // TDM-stage 16 phi rows into LDS (one DMA descriptor, wave 0 issues)
    if (wave == 0) {
        tdm_load_tile_bf16(0u, phi + (size_t)mbase * K2, K2, 16, K2);
        __builtin_amdgcn_s_wait_tensorcnt(0);
    }
    // Keep the DMA-written tile visible to alias analysis (see k_gemm1).
    if (Ct == nullptr) sPhi[tid] = (__bf16)0.0f;
    asm volatile("" : : "v"(sPhi) : "memory");
    __syncthreads();

    int lrow = lane & 15;
    int hi = lane >> 4;
    int ksel = hi << 4;
    float rn[8];
#pragma unroll
    for (int i = 0; i < 8; ++i) rn[i] = rownorm[mbase + hi * 8 + i];
    float v0[8], v1[8], v2[8];
#pragma unroll
    for (int i = 0; i < 8; ++i) { v0[i] = 3.0e38f; v1[i] = 3.0e38f; v2[i] = 3.0e38f; }

    const __bf16* arow = sPhi + lrow * K2 + ksel;
    for (int nt = wave; nt < 196; nt += 4) {
        const __bf16* crow = Ct + (size_t)(nt * 16 + lrow) * K2 + ksel;
        if (nt + 4 < 196)   // warm the next N-tile's rows into cache
            __builtin_prefetch(Ct + (size_t)((nt + 4) * 16 + lrow) * K2 + ksel, 0, 1);
        v8f acc = {};
#pragma unroll 4
        for (int kc = 0; kc < 56; ++kc) {
            v16bf a = *(const v16bf*)(arow + kc * 32);
            v16bf bm = *(const v16bf*)(crow + kc * 32);
            acc = __builtin_amdgcn_wmma_f32_16x16x32_bf16(false, a, false, bm,
                                                          (short)0, acc, false, false);
        }
        float cn = colnorm[nt * 16 + lrow];
#pragma unroll
        for (int i = 0; i < 8; ++i) {
            float d2 = rn[i] + cn - 2.0f * acc[i];
            float d = sqrtf(fmaxf(d2, 0.0f));
            ins3(d, v0[i], v1[i], v2[i]);
        }
    }

    // merge sorted triples across the 16-lane half (rows are per-slot invariant)
    for (int m = 1; m < 16; m <<= 1) {
#pragma unroll
        for (int i = 0; i < 8; ++i) {
            float b0 = __shfl_xor(v0[i], m, 32);
            float b1 = __shfl_xor(v1[i], m, 32);
            float b2 = __shfl_xor(v2[i], m, 32);
            merge3(v0[i], v1[i], v2[i], b0, b1, b2);
        }
    }
    if (lrow == 0) {
#pragma unroll
        for (int i = 0; i < 8; ++i) {
            int r = hi * 8 + i;
            sRed[wave][r][0] = v0[i];
            sRed[wave][r][1] = v1[i];
            sRed[wave][r][2] = v2[i];
        }
    }
    __syncthreads();
    if (tid < 16) {
        float a0 = sRed[0][tid][0], a1 = sRed[0][tid][1], a2 = sRed[0][tid][2];
#pragma unroll
        for (int w = 1; w < 4; ++w)
            merge3(a0, a1, a2, sRed[w][tid][0], sRed[w][tid][1], sRed[w][tid][2]);
        // softmin over (a0,a1,a2); score = w0 * a0
        float e1 = __expf(a0 - a1);
        float e2 = __expf(a0 - a2);
        float w0 = 1.0f / (1.0f + e1 + e2);
        out[mbase + tid] = w0 * a0;
    }
}

// ============================================================================
extern "C" void kernel_launch(void* const* d_in, const int* in_sizes, int n_in,
                              void* d_out, int out_size, void* d_ws, size_t ws_size,
                              hipStream_t stream) {
    (void)in_sizes; (void)n_in; (void)out_size; (void)ws_size;
    const float* p1 = (const float*)d_in[0];   // (8,256,56,56)
    const float* p2 = (const float*)d_in[1];   // (8,512,28,28)
    const float* p3 = (const float*)d_in[2];   // (8,1024,14,14)
    const float* W  = (const float*)d_in[3];   // (1792,1794)
    const float* b  = (const float*)d_in[4];   // (1792,)
    const float* C  = (const float*)d_in[5];   // (1792,3136)
    float* out = (float*)d_out;

    char* ws = (char*)d_ws;
    __bf16* Wb   = (__bf16*)(ws + OFF_WB);
    __bf16* Ct   = (__bf16*)(ws + OFF_CT);
    float*  cn   = (float*)(ws + OFF_CN);
    float*  ap1  = (float*)(ws + OFF_AP1);
    float*  ap2  = (float*)(ws + OFF_AP2);
    float*  ap3  = (float*)(ws + OFF_AP3);
    __bf16* Xb   = (__bf16*)(ws + OFF_X);
    __bf16* phi  = (__bf16*)(ws + OFF_PHI);
    float*  rn   = (float*)(ws + OFF_RN);

    // weight/centroid prep
    k_prep_W<<<(NPHI * K1PAD + 255) / 256, 256, 0, stream>>>(W, Wb);
    k_transpose_C<<<dim3(NCENT / 32, K2 / 32), 256, 0, stream>>>(C, Ct);
    k_colnorm<<<(NCENT + 255) / 256, 256, 0, stream>>>(C, cn);

    // descriptor: pool each level, assemble bf16 X with coords + K padding
    k_avgpool<<<(BATCH * 256 * 56 * 56 + 255) / 256, 256, 0, stream>>>(p1, ap1, 256, 56);
    k_avgpool<<<(BATCH * 512 * 28 * 28 + 255) / 256, 256, 0, stream>>>(p2, ap2, 512, 28);
    k_avgpool<<<(BATCH * 1024 * 14 * 14 + 255) / 256, 256, 0, stream>>>(p3, ap3, 1024, 14);
    k_build_X<<<(MROWS * K1PAD + 255) / 256, 256, 0, stream>>>(ap1, ap2, ap3, Xb);

    // GEMM1 (bf16 WMMA, TDM-staged A) -> phi, then row norms
    k_gemm1<<<MROWS / 16, 128, 0, stream>>>(Xb, Wb, b, phi);
    k_rownorm<<<MROWS / 4, 128, 0, stream>>>(phi, rn);

    // GEMM2 (bf16 WMMA, TDM-staged A) fused with distance/top-3/softmin
    k_gemm2_topk<<<MROWS / 16, 128, 0, stream>>>(phi, Ct, rn, cn, out);
}
